// _GCNEncoder_2757369004598
// MI455X (gfx1250) — matrix operand order repaired and checked by reading, not compiled, not run on hardware
//
#include <hip/hip_runtime.h>
#include <hip/hip_bf16.h>

#define N_NODES 50000
#define N_EDGES 800000
#define IN_C    256
#define HID_C   256
#define OUT_C   128
#define KDIM    256                 // both GEMMs have K = 256
#define BPAD    8                   // LDS pad (elements) to break bank conflicts

typedef __attribute__((ext_vector_type(16))) __bf16 v16bf;
typedef __attribute__((ext_vector_type(8)))  float  v8f;

union BFrag  { v16bf v; uint4 q[2]; __bf16 h[16]; };
union BF8    { __bf16 h[8]; uint4 v; };

// f32 -> bf16 through the HIP intrinsic (lowers to HW cvt); memcpy dodges
// any struct-vs-native __hip_bfloat16 definition differences.
__device__ __forceinline__ __bf16 f2bf16(float f) {
    __hip_bfloat16 t = __float2bfloat16(f);
    __bf16 r;
    __builtin_memcpy(&r, &t, 2);
    return r;
}

// -------------------------------------------------------------------------
// One-shot f32 -> bf16 conversion (x, W1, W2), 8 elements/thread, 128b stores
// -------------------------------------------------------------------------
__global__ void cvt_f32_bf16(const float* __restrict__ in, __bf16* __restrict__ out,
                             long long n8)
{
    long long i = (long long)blockIdx.x * blockDim.x + threadIdx.x;
    if (i >= n8) return;
    const float4* p = reinterpret_cast<const float4*>(in + i * 8);
    float4 a = p[0], b = p[1];
    BF8 o;
    o.h[0] = f2bf16(a.x); o.h[1] = f2bf16(a.y); o.h[2] = f2bf16(a.z); o.h[3] = f2bf16(a.w);
    o.h[4] = f2bf16(b.x); o.h[5] = f2bf16(b.y); o.h[6] = f2bf16(b.z); o.h[7] = f2bf16(b.w);
    reinterpret_cast<uint4*>(out)[i] = o.v;
}

// ReLU fused with bf16 down-conversion (feeds GEMM2 directly)
__global__ void relu_cvt_bf16(const float* __restrict__ in, __bf16* __restrict__ out,
                              long long n8)
{
    long long i = (long long)blockIdx.x * blockDim.x + threadIdx.x;
    if (i >= n8) return;
    const float4* p = reinterpret_cast<const float4*>(in + i * 8);
    float4 a = p[0], b = p[1];
    BF8 o;
    o.h[0] = f2bf16(fmaxf(a.x, 0.f)); o.h[1] = f2bf16(fmaxf(a.y, 0.f));
    o.h[2] = f2bf16(fmaxf(a.z, 0.f)); o.h[3] = f2bf16(fmaxf(a.w, 0.f));
    o.h[4] = f2bf16(fmaxf(b.x, 0.f)); o.h[5] = f2bf16(fmaxf(b.y, 0.f));
    o.h[6] = f2bf16(fmaxf(b.z, 0.f)); o.h[7] = f2bf16(fmaxf(b.w, 0.f));
    reinterpret_cast<uint4*>(out)[i] = o.v;
}

// -------------------------------------------------------------------------
// WMMA GEMM: C[M,N] = A[M,256] * B[256,N], bf16 inputs, f32 accumulate.
// 8 waves/block, one 16x16 tile per wave; B tile (256x16) staged once per
// block into LDS transposed [n][k] (+pad), so fragments are two ds_load_b128.
// Fragment lane layouts per CDNA5 ISA:
//   A 16x32: lanes 0-15 row m0+l, K{0..7,16..23}; lanes 16-31 K{8..15,24..31}
//   B 32x16: lanes 0-15 col n0+l, K{0..7,16..23}; lanes 16-31 K{8..15,24..31}
//   C/D:     VGPR r -> row m0+r / m0+8+r, col n0+l
// -------------------------------------------------------------------------
__global__ void gemm_bf16_wmma(const __bf16* __restrict__ A,
                               const __bf16* __restrict__ B,
                               float* __restrict__ C,
                               int M, int N)
{
    __shared__ __align__(16) __bf16 sBT[16 * (KDIM + BPAD)];

    const int t    = threadIdx.x;       // 0..255
    const int lane = t & 31;
    const int wave = t >> 5;
    const int l15  = lane & 15;
    const int hi   = lane >> 4;
    const int n0   = blockIdx.y * 16;

    // ---- stage B[:, n0:n0+16] transposed into LDS (thread t = row k=t) ----
    {
        BFrag row;
        const uint4* bp = reinterpret_cast<const uint4*>(B + (size_t)t * N + n0);
        row.q[0] = bp[0];
        row.q[1] = bp[1];
        #pragma unroll
        for (int j = 0; j < 16; ++j)
            sBT[j * (KDIM + BPAD) + t] = row.h[j];
    }
    __syncthreads();

    const int m0 = (blockIdx.x * 8 + wave) * 16;
    if (m0 < M) {                       // wave-uniform: EXEC all-ones inside
        v8f acc = {0.f, 0.f, 0.f, 0.f, 0.f, 0.f, 0.f, 0.f};
        const __bf16* arow = A + (size_t)(m0 + l15) * KDIM;

        #pragma unroll
        for (int k0 = 0; k0 < KDIM; k0 += 32) {
            BFrag af, bf;
            // A fragment: two contiguous 8-elem (16B) chunks
            af.q[0] = *reinterpret_cast<const uint4*>(arow + k0 + hi * 8);
            af.q[1] = *reinterpret_cast<const uint4*>(arow + k0 + hi * 8 + 16);
            // B fragment: two 8-elem chunks matching A's K order
            const __bf16* bp = &sBT[l15 * (KDIM + BPAD) + k0 + hi * 8];
            bf.q[0] = *reinterpret_cast<const uint4*>(bp);
            bf.q[1] = *reinterpret_cast<const uint4*>(bp + 16);

            acc = __builtin_amdgcn_wmma_f32_16x16x32_bf16(
                /*neg_a=*/false, af.v, /*neg_b=*/false, bf.v,
                /*c_mod=*/(short)0, acc, /*reuse_a=*/false, /*reuse_b=*/false);
        }

        float* crow = C + (size_t)(m0 + hi * 8) * N + (n0 + l15);
        #pragma unroll
        for (int r = 0; r < 8; ++r)
            crow[(size_t)r * N] = acc[r];
    }
}

// -------------------------------------------------------------------------
// Degree / normalization
// -------------------------------------------------------------------------
__global__ void deg_init(float* deg, int n) {
    int i = blockIdx.x * blockDim.x + threadIdx.x;
    if (i < n) deg[i] = 1.0f;                       // self-loop
}

__global__ void deg_count(const int* __restrict__ dst, float* deg, int nE) {
    int e = blockIdx.x * blockDim.x + threadIdx.x;
    if (e < nE) atomicAdd(&deg[dst[e]], 1.0f);
}

__global__ void make_dinv(const float* __restrict__ deg, float* dinv, int n) {
    int i = blockIdx.x * blockDim.x + threadIdx.x;
    if (i < n) dinv[i] = rsqrtf(deg[i]);
}

// out[i,c] = t[i,c]*dinv[i]^2 + bias[c]  (self-loop msg + bias; full overwrite)
__global__ void init_agg(const float* __restrict__ t, const float* __restrict__ dinv,
                         const float* __restrict__ bias, float* __restrict__ out, int C)
{
    int i = blockIdx.x;
    int c = threadIdx.x;
    float di = dinv[i];
    out[(size_t)i * C + c] = t[(size_t)i * C + c] * di * di + bias[c];
}

// One block per edge, one lane per channel: coalesced row gather + f32 atomics
// that resolve in the 192MB L2 (51MB destination) — the bandwidth roofline.
__global__ void scatter_edges(const float* __restrict__ t, const float* __restrict__ dinv,
                              const int* __restrict__ src, const int* __restrict__ dst,
                              float* __restrict__ out, int C)
{
    int e = blockIdx.x;
    int c = threadIdx.x;
    int s = src[e];
    int d = dst[e];
    float norm = dinv[s] * dinv[d];
    atomicAdd(&out[(size_t)d * C + c], t[(size_t)s * C + c] * norm);
}

// -------------------------------------------------------------------------
// Launch
// -------------------------------------------------------------------------
extern "C" void kernel_launch(void* const* d_in, const int* in_sizes, int n_in,
                              void* d_out, int out_size, void* d_ws, size_t ws_size,
                              hipStream_t stream)
{
    const float* x   = (const float*)d_in[0];       // [50000,256]
    const int*   ei  = (const int*)  d_in[1];       // [2,800000]
    const float* W1  = (const float*)d_in[2];       // [256,256]
    const float* b1  = (const float*)d_in[3];       // [256]
    const float* W2  = (const float*)d_in[4];       // [256,128]
    const float* b2  = (const float*)d_in[5];       // [128]
    float*       out = (float*)d_out;               // [50000,128]

    const int* src = ei;
    const int* dst = ei + N_EDGES;

    // Workspace layout (~180 MB)
    char* ws = (char*)d_ws;
    float*  t1   = (float*)ws;                               ws += sizeof(float) * (size_t)N_NODES * HID_C;
    float*  a1   = (float*)ws;                               ws += sizeof(float) * (size_t)N_NODES * HID_C;
    float*  t2   = (float*)ws;                               ws += sizeof(float) * (size_t)N_NODES * OUT_C;
    __bf16* xbf  = (__bf16*)ws;                              ws += 2 * (size_t)N_NODES * IN_C;
    __bf16* a1bf = (__bf16*)ws;                              ws += 2 * (size_t)N_NODES * HID_C;
    __bf16* w1bf = (__bf16*)ws;                              ws += 2 * (size_t)IN_C * HID_C;
    __bf16* w2bf = (__bf16*)ws;                              ws += 2 * (size_t)HID_C * OUT_C;
    float*  deg  = (float*)ws;                               ws += sizeof(float) * N_NODES;
    float*  dinv = (float*)ws;

    // one-shot bf16 conversions of GEMM operands
    long long nx8 = (long long)N_NODES * IN_C / 8;
    cvt_f32_bf16<<<(int)((nx8 + 255) / 256), 256, 0, stream>>>(x,  xbf,  nx8);
    cvt_f32_bf16<<<(IN_C * HID_C / 8 + 255) / 256, 256, 0, stream>>>(W1, w1bf, IN_C * HID_C / 8);
    cvt_f32_bf16<<<(HID_C * OUT_C / 8 + 255) / 256, 256, 0, stream>>>(W2, w2bf, HID_C * OUT_C / 8);

    // degrees (incl. self-loops) -> dinv
    deg_init <<<(N_NODES + 255) / 256, 256, 0, stream>>>(deg, N_NODES);
    deg_count<<<(N_EDGES + 255) / 256, 256, 0, stream>>>(dst, deg, N_EDGES);
    make_dinv<<<(N_NODES + 255) / 256, 256, 0, stream>>>(deg, dinv, N_NODES);

    // Layer 1: t1 = x @ W1 ; a1 = agg(t1) + b1 ; a1bf = bf16(relu(a1))
    dim3 g1((N_NODES / 16 + 7) / 8, HID_C / 16);
    gemm_bf16_wmma<<<g1, 256, 0, stream>>>(xbf, w1bf, t1, N_NODES, HID_C);
    init_agg      <<<N_NODES, HID_C, 0, stream>>>(t1, dinv, b1, a1, HID_C);
    scatter_edges <<<N_EDGES, HID_C, 0, stream>>>(t1, dinv, src, dst, a1, HID_C);
    long long nh8 = (long long)N_NODES * HID_C / 8;
    relu_cvt_bf16 <<<(int)((nh8 + 255) / 256), 256, 0, stream>>>(a1, a1bf, nh8);

    // Layer 2: t2 = relu(a1) @ W2 ; out = agg(t2) + b2
    dim3 g2((N_NODES / 16 + 7) / 8, OUT_C / 16);
    gemm_bf16_wmma<<<g2, 256, 0, stream>>>(a1bf, w2bf, t2, N_NODES, OUT_C);
    init_agg      <<<N_NODES, OUT_C, 0, stream>>>(t2, dinv, b2, out, OUT_C);
    scatter_edges <<<N_EDGES, OUT_C, 0, stream>>>(t2, dinv, src, dst, out, OUT_C);
}
